// ConvRefinerWithCorr_74680891342865
// MI455X (gfx1250) — matrix operand order repaired and checked by reading, not compiled, not run on hardware
//
#include <hip/hip_runtime.h>
#include <hip/hip_bf16.h>

typedef _Float16 half_t;
typedef __attribute__((ext_vector_type(16))) _Float16 v16h;
typedef __attribute__((ext_vector_type(8)))  _Float16 v8h;
typedef __attribute__((ext_vector_type(8)))  float    v8f;

#define BB    2
#define CC    256
#define HH    64
#define WW    64
#define NPIX  (BB*HH*WW)
#define K0    640          // 625 real input channels, padded to 640 (mult of 32)
#define IN_DIM 625
#define NH    512
#define KTAPS 49

// ---------------------------------------------------------------- helpers
__device__ __forceinline__ float loadImg(const float* __restrict__ p,
                                         int b, int c, int yy, int xx) {
  if (xx < 0 || xx >= WW || yy < 0 || yy >= HH) return 0.f;   // zero-pad sampling
  return p[(size_t)(b*CC + c)*(HH*WW) + yy*WW + xx];
}

__device__ __forceinline__ v8h zero_v8h() {
  v8h z;
#pragma unroll
  for (int i = 0; i < 8; ++i) z[i] = (half_t)0.f;
  return z;
}

// ---------------------------------------------------------------- prep: x, x_hat, emb
__global__ __launch_bounds__(256) void prep_main(
    const float* __restrict__ x, const float* __restrict__ y,
    const float* __restrict__ flow,
    const float* __restrict__ de_w, const float* __restrict__ de_b,
    half_t* __restrict__ d0)
{
  int bh = blockIdx.x;                 // 0 .. B*H-1
  int b = bh / HH, h = bh % HH;
  int t = threadIdx.x;
  int w  = t & 63;
  int lc = t >> 6;                     // 0..3 channel groups
  int pix = bh*WW + w;
  half_t* row = d0 + (size_t)pix*K0;

  float fx = flow[(size_t)(b*2+0)*(HH*WW) + h*WW + w];
  float fy = flow[(size_t)(b*2+1)*(HH*WW) + h*WW + w];
  float X = (fx + 1.f)*32.f - 0.5f;
  float Y = (fy + 1.f)*32.f - 0.5f;
  float xf = floorf(X), yf = floorf(Y);
  int x0 = (int)xf, y0 = (int)yf;
  float wx1 = X - xf, wy1 = Y - yf;
  float wx0 = 1.f - wx1, wy0 = 1.f - wy1;
  float w00 = wy0*wx0, w01 = wy0*wx1, w10 = wy1*wx0, w11 = wy1*wx1;

  for (int c0 = 0; c0 < 64; ++c0) {
    int c = c0*4 + lc;
    float xv = x[(size_t)(b*CC + c)*(HH*WW) + h*WW + w];
    row[c] = (half_t)xv;                                   // cols 0..255 : x
    float v00 = loadImg(y,b,c,y0,  x0);
    float v01 = loadImg(y,b,c,y0,  x0+1);
    float v10 = loadImg(y,b,c,y0+1,x0);
    float v11 = loadImg(y,b,c,y0+1,x0+1);
    row[CC + c] = (half_t)(w00*v00 + w01*v01 + w10*v10 + w11*v11);  // 256..511 : x_hat
  }
  // emb: cols 512..575, 40/32 * (flow - identity_grid)
  float ax = -1.f + (2.f*w + 1.f)/(float)WW;
  float ay = -1.f + (2.f*h + 1.f)/(float)HH;
  float ix = 1.25f*(fx - ax);
  float iy = 1.25f*(fy - ay);
  for (int j = 0; j < 16; ++j) {
    int d = lc*16 + j;
    float e = de_w[d*2+0]*ix + de_w[d*2+1]*iy + de_b[d];
    row[2*CC + d] = (half_t)e;
  }
}

// ---------------------------------------------------------------- prep: local correlation (49 taps)
__global__ __launch_bounds__(64) void prep_corr(
    const float* __restrict__ x, const float* __restrict__ y,
    const float* __restrict__ flow, half_t* __restrict__ d0)
{
  int pix = blockIdx.x;
  int b = pix >> 12;
  int sp = pix & 4095;
  int k = threadIdx.x;
  half_t* row = d0 + (size_t)pix*K0 + 2*CC + 64;           // col 576
  if (k >= KTAPS) { if (k < 64) row[k] = (half_t)0.f; return; }  // zero pad cols 625..639

  float fx = flow[(size_t)(b*2+0)*(HH*WW) + sp];
  float fy = flow[(size_t)(b*2+1)*(HH*WW) + sp];
  float X = (fx + 1.f)*32.f - 0.5f;
  float Y = (fy + 1.f)*32.f - 0.5f;
  float xf = floorf(X), yf = floorf(Y);
  float wx1 = X - xf, wy1 = Y - yf;
  float wx0 = 1.f - wx1, wy0 = 1.f - wy1;
  // window step in normalized coords == exactly 1 pixel -> integer offsets, shared fraction
  int dx = (k % 7) - 3, dy = (k / 7) - 3;
  int x0 = (int)xf + dx, y0 = (int)yf + dy;
  float w00 = wy0*wx0, w01 = wy0*wx1, w10 = wy1*wx0, w11 = wy1*wx1;

  float acc = 0.f;
  for (int c = 0; c < CC; ++c) {
    float f0 = x[(size_t)(b*CC + c)*(HH*WW) + sp];
    float s = w00*loadImg(y,b,c,y0,  x0) + w01*loadImg(y,b,c,y0,  x0+1)
            + w10*loadImg(y,b,c,y0+1,x0) + w11*loadImg(y,b,c,y0+1,x0+1);
    acc += f0 * s;
  }
  row[k] = (half_t)(acc * 0.0625f);    // 1/sqrt(256)
}

// ---------------------------------------------------------------- weight packing
// src fp32 [N][Cin][5][5]  ->  dst fp16 [25][N][Kc] (Kc >= Cin, zero padded)
__global__ __launch_bounds__(256) void pack_w5(
    const float* __restrict__ src, half_t* __restrict__ dst,
    int N, int Cin, int Kc, size_t total)
{
  size_t idx = (size_t)blockIdx.x*256 + threadIdx.x;
  if (idx >= total) return;
  int k  = (int)(idx % (size_t)Kc);
  int oc = (int)((idx / (size_t)Kc) % (size_t)N);
  int tp = (int)(idx / ((size_t)Kc * (size_t)N));
  float v = (k < Cin) ? src[((size_t)oc*Cin + k)*25 + tp] : 0.f;
  dst[idx] = (half_t)v;
}

__global__ __launch_bounds__(256) void pack_cast(
    const float* __restrict__ src, half_t* __restrict__ dst, size_t total)
{
  size_t idx = (size_t)blockIdx.x*256 + threadIdx.x;
  if (idx < total) dst[idx] = (half_t)src[idx];
}

// ---------------------------------------------------------------- WMMA implicit-GEMM conv
// act : fp16 [B*H*W][Kc]   (channels-last)
// wgt : fp16 [ntaps][512][Kc]
// out : fp16 [B*H*W][512]
// ntaps==25 -> 5x5 conv (pad 2) ; ntaps==1 -> 1x1 conv
// Workgroup tile: 64 pixels x 256 ocs; wave tile: 32x64 (2x4 WMMA C blocks).
// Software pipelined: next (tap,k) tile prefetched into regs during WMMA burst.
__global__ __launch_bounds__(256) void gemm_conv(
    const half_t* __restrict__ act, int Kc,
    const half_t* __restrict__ wgt, int ntaps,
    const float* __restrict__ bias,
    const float* __restrict__ g,  const float* __restrict__ beta,
    const float* __restrict__ mu, const float* __restrict__ var,
    int do_bn, half_t* __restrict__ out)
{
  constexpr int LDP = 40;   // padded LDS row stride (halves): 80B = 20 banks, conflict-free
  __shared__ __align__(16) half_t ldsA[64*LDP];     //  5.0 KB
  __shared__ __align__(16) half_t ldsB[256*LDP];    // 20.0 KB

  const int t    = threadIdx.x;
  const int lane = t & 31;
  const int wave = t >> 5;
  const int mb = wave & 1;             // 0..1 : 32-row block
  const int nb = wave >> 1;            // 0..3 : 64-col block
  const int n_base = blockIdx.x * 256;
  const int bh = blockIdx.y;           // one image row per block
  const int b = bh / HH, h = bh % HH;
  const int pix_base = bh * WW;

  // cooperative-load mappings
  const int rA   = t >> 2;             // 0..63 pixel within row
  const int segA = t & 3;              // 8-half segment of 32
  const int ocB  = t;                  // 0..255 : each thread stages one full B row

  const int hs  = lane >> 4;           // half-wave select
  const int l16 = lane & 15;

  v8f acc[2][4];
#pragma unroll
  for (int i = 0; i < 2; ++i)
#pragma unroll
    for (int j = 0; j < 4; ++j)
#pragma unroll
      for (int e = 0; e < 8; ++e) acc[i][j][e] = 0.f;

  const int S = ntaps * (Kc >> 5);

  // ---- register prefetch state (next tile to be stored to LDS)
  int tapN = 0, kkN = 0;
  v8h avp;
  v8h bvp[4];

  auto fetch = [&](int tap, int kk) {
    int dy = 0, dx = 0;
    if (ntaps > 1) { dy = tap/5 - 2; dx = tap%5 - 2; }
    int h_in = h + dy;
    int w_in = rA + dx;
    bool vAr = (h_in >= 0 && h_in < HH && w_in >= 0 && w_in < WW);
    avp = zero_v8h();
    if (vAr)
      avp = *(const v8h*)(act + (size_t)((b*HH + h_in)*WW + w_in)*Kc + kk + segA*8);
    const half_t* wr = wgt + ((size_t)tap*NH + n_base + ocB)*Kc + kk;
#pragma unroll
    for (int q = 0; q < 4; ++q) bvp[q] = *(const v8h*)(wr + q*8);
  };

  fetch(0, 0);

  for (int s = 0; s < S; ++s) {
    __syncthreads();
    // commit prefetched tile to LDS
    *(v8h*)(ldsA + rA*LDP + segA*8) = avp;
#pragma unroll
    for (int q = 0; q < 4; ++q) *(v8h*)(ldsB + ocB*LDP + q*8) = bvp[q];
    __syncthreads();

    // issue next tile's global loads; they retire while WMMAs run
    kkN += 32;
    if (kkN == Kc) { kkN = 0; ++tapN; }
    if (s + 1 < S) fetch(tapN, kkN);

    // A fragments: lane<16 -> row l16, k {0..7,16..23}; lane>=16 -> k {8..15,24..31}
    v16h afrag[2], bfrag[4];
#pragma unroll
    for (int im = 0; im < 2; ++im) {
      int m = mb*32 + im*16 + l16;
      v8h lo = *(const v8h*)(ldsA + m*LDP + hs*8);
      v8h hi = *(const v8h*)(ldsA + m*LDP + 16 + hs*8);
#pragma unroll
      for (int e = 0; e < 8; ++e) { afrag[im][e] = lo[e]; afrag[im][8+e] = hi[e]; }
    }
  // B fragments: lane holds col l16, k {hs*8..hs*8+7, 16+hs*8..16+hs*8+7}
#pragma unroll
    for (int jn = 0; jn < 4; ++jn) {
      int oc = nb*64 + jn*16 + l16;
      v8h lo = *(const v8h*)(ldsB + oc*LDP + hs*8);
      v8h hi = *(const v8h*)(ldsB + oc*LDP + 16 + hs*8);
#pragma unroll
      for (int e = 0; e < 8; ++e) { bfrag[jn][e] = lo[e]; bfrag[jn][8+e] = hi[e]; }
    }
#pragma unroll
    for (int im = 0; im < 2; ++im)
#pragma unroll
      for (int jn = 0; jn < 4; ++jn)
        acc[im][jn] = __builtin_amdgcn_wmma_f32_16x16x32_f16(
            false, afrag[im], false, bfrag[jn],
            (short)0, acc[im][jn], false, false);
  }

  // epilogue: bias (+ BN + ReLU), store fp16
#pragma unroll
  for (int im = 0; im < 2; ++im) {
    int m0 = mb*32 + im*16;
#pragma unroll
    for (int jn = 0; jn < 4; ++jn) {
      int oc = n_base + nb*64 + jn*16 + l16;
      float bv = bias[oc];
      float sc = 1.f, sh = 0.f;
      if (do_bn) {
        sc = g[oc] * rsqrtf(var[oc] + 1e-5f);
        sh = beta[oc] - mu[oc]*sc;
      }
#pragma unroll
      for (int e = 0; e < 8; ++e) {
        float v = acc[im][jn][e] + bv;
        if (do_bn) { v = v*sc + sh; v = v > 0.f ? v : 0.f; }
        int row = m0 + hs*8 + e;                       // D layout: M = e + 8*hs
        out[(size_t)(pix_base + row)*NH + oc] = (half_t)v;
      }
    }
  }
}

// ---------------------------------------------------------------- output head (512 -> 3, fp32 out)
__global__ __launch_bounds__(256) void out_conv(
    const half_t* __restrict__ act, const float* __restrict__ ow,
    const float* __restrict__ ob, float* __restrict__ outp)
{
  int idx = blockIdx.x*256 + threadIdx.x;
  if (idx >= NPIX*3) return;
  int oc  = idx % 3;
  int pix = idx / 3;
  const half_t* a = act + (size_t)pix*NH;
  const float* wr = ow + oc*NH;
  float s = ob[oc];
  for (int k = 0; k < NH; ++k) s += (float)a[k] * wr[k];
  int b = pix >> 12, sp = pix & 4095;
  if (oc < 2) outp[(size_t)(b*2 + oc)*4096 + sp] = s;          // displacement
  else        outp[(size_t)16384 + (size_t)b*4096 + sp] = s;   // certainty
}

// ---------------------------------------------------------------- launch
extern "C" void kernel_launch(void* const* d_in, const int* in_sizes, int n_in,
                              void* d_out, int out_size, void* d_ws, size_t ws_size,
                              hipStream_t stream)
{
  const float* x      = (const float*)d_in[0];
  const float* y      = (const float*)d_in[1];
  const float* flow   = (const float*)d_in[2];
  const float* de_w   = (const float*)d_in[3];
  const float* de_b   = (const float*)d_in[4];
  const float* b1_w1  = (const float*)d_in[5];
  const float* b1_b1  = (const float*)d_in[6];
  const float* b1_g   = (const float*)d_in[7];
  const float* b1_beta= (const float*)d_in[8];
  const float* b1_mu  = (const float*)d_in[9];
  const float* b1_var = (const float*)d_in[10];
  const float* b1_w2  = (const float*)d_in[11];
  const float* b1_b2  = (const float*)d_in[12];
  const float* h_w1   = (const float*)d_in[13];
  const float* h_b1   = (const float*)d_in[14];
  const float* h_g    = (const float*)d_in[15];
  const float* h_beta = (const float*)d_in[16];
  const float* h_mu   = (const float*)d_in[17];
  const float* h_var  = (const float*)d_in[18];
  const float* h_w2   = (const float*)d_in[19];
  const float* h_b2   = (const float*)d_in[20];
  const float* out_w  = (const float*)d_in[21];
  const float* out_b  = (const float*)d_in[22];
  (void)in_sizes; (void)n_in; (void)out_size; (void)ws_size;

  char* ws = (char*)d_ws;
  size_t off = 0;
  auto take = [&](size_t bytes) -> char* {
    char* r = ws + off;
    off = (off + bytes + 255) & ~(size_t)255;
    return r;
  };
  half_t* d0   = (half_t*)take((size_t)NPIX*K0*2);        // 10.5 MB  concat input
  half_t* actA = (half_t*)take((size_t)NPIX*NH*2);        // 8.4 MB
  half_t* actB = (half_t*)take((size_t)NPIX*NH*2);        // 8.4 MB
  half_t* W5a  = (half_t*)take((size_t)25*NH*K0*2);       // 16.4 MB  block1 5x5
  half_t* W5h  = (half_t*)take((size_t)3*25*NH*NH*2);     // 39.3 MB  hidden 5x5
  half_t* W1a  = (half_t*)take((size_t)NH*NH*2);          // block1 1x1
  half_t* W1h  = (half_t*)take((size_t)3*NH*NH*2);        // hidden 1x1

  // ---- pack weights to fp16 [tap][oc][k]
  {
    size_t tot = (size_t)25*NH*K0;
    pack_w5<<<(unsigned)((tot+255)/256), 256, 0, stream>>>(b1_w1, W5a, NH, IN_DIM, K0, tot);
  }
  for (int i = 0; i < 3; ++i) {
    size_t tot = (size_t)25*NH*NH;
    pack_w5<<<(unsigned)((tot+255)/256), 256, 0, stream>>>(
        h_w1 + (size_t)i*NH*NH*25, W5h + (size_t)i*25*NH*NH, NH, NH, NH, tot);
  }
  {
    size_t tot = (size_t)NH*NH;
    pack_cast<<<(unsigned)((tot+255)/256), 256, 0, stream>>>(b1_w2, W1a, tot);
  }
  {
    size_t tot = (size_t)3*NH*NH;
    pack_cast<<<(unsigned)((tot+255)/256), 256, 0, stream>>>(h_w2, W1h, tot);
  }

  // ---- build concat input d0 = [x | x_hat | emb | lc | pad]
  prep_main<<<BB*HH, 256, 0, stream>>>(x, y, flow, de_w, de_b, d0);
  prep_corr<<<NPIX, 64, 0, stream>>>(x, y, flow, d0);

  // ---- conv stack (all WMMA implicit GEMM)
  dim3 ggrid(NH/256, BB*HH);
  gemm_conv<<<ggrid, 256, 0, stream>>>(d0, K0, W5a, 25,
      b1_b1, b1_g, b1_beta, b1_mu, b1_var, 1, actA);
  gemm_conv<<<ggrid, 256, 0, stream>>>(actA, NH, W1a, 1,
      b1_b2, nullptr, nullptr, nullptr, nullptr, 0, actB);
  for (int i = 0; i < 3; ++i) {
    gemm_conv<<<ggrid, 256, 0, stream>>>(actB, NH, W5h + (size_t)i*25*NH*NH, 25,
        h_b1 + i*NH, h_g + i*NH, h_beta + i*NH, h_mu + i*NH, h_var + i*NH, 1, actA);
    gemm_conv<<<ggrid, 256, 0, stream>>>(actA, NH, W1h + (size_t)i*NH*NH, 1,
        h_b2 + i*NH, nullptr, nullptr, nullptr, nullptr, 0, actB);
  }

  // ---- output head
  out_conv<<<(NPIX*3 + 255)/256, 256, 0, stream>>>(actB, out_w, out_b, (float*)d_out);
}
